// DRL_head_49143015801417
// MI455X (gfx1250) — compile-verified, matrix operand order checked by reading, not run on hardware
//
#include <hip/hip_runtime.h>

// ---------------------------------------------------------------------------
// Fused DRL head for MI455X (gfx1250, wave32).
//  Pass 1: L2-normalize rows of text/video, convert to bf16 into d_ws.
//  Pass 2: per (a, b) pair, compute the 32x16 logit tile with
//          v_wmma_f32_16x16x32_bf16 (fp32 accum), then do both softmax-pooling
//          levels in-registers with wave32 shuffles. Logits never touch HBM.
// ---------------------------------------------------------------------------

typedef __attribute__((ext_vector_type(16))) __bf16 v16bf;
typedef __attribute__((ext_vector_type(8)))  float  v8f;

#define A_N   256
#define T_TOK 32
#define B_N   256
#define V_FRM 16
#define DD    512
#define TAUF  100.0f
#define STR   520   // LDS row stride in halfs: 512 + 8 pad -> 1040B, banks rotate 4/row

__device__ __forceinline__ float xsh(float v, int m) { return __shfl_xor(v, m, 32); }

__device__ __forceinline__ unsigned short f2bf(float f) {
    unsigned u = __float_as_uint(f);
    unsigned r = (u + 0x7FFFu + ((u >> 16) & 1u)) >> 16;   // round-to-nearest-even
    return (unsigned short)r;
}

// ------------------------- Pass 1: normalize + cvt -------------------------
// One wave per 512-float row; lane reads 4 float4s, wave-reduces sum of
// squares, writes 512 bf16 (as uint2 pairs).
__global__ __launch_bounds__(256) void drl_norm_bf16(const float* __restrict__ in,
                                                     unsigned short* __restrict__ outb,
                                                     int nrows) {
    int wave = threadIdx.x >> 5, lane = threadIdx.x & 31;
    int row  = blockIdx.x * 8 + wave;
    if (row >= nrows) return;
    const float4* r = (const float4*)(in + (size_t)row * DD);
    float4 x[4];
    float ss = 0.f;
#pragma unroll
    for (int c = 0; c < 4; ++c) {
        x[c] = r[lane + 32 * c];
        ss += x[c].x * x[c].x + x[c].y * x[c].y + x[c].z * x[c].z + x[c].w * x[c].w;
    }
#pragma unroll
    for (int m = 1; m < 32; m <<= 1) ss += xsh(ss, m);
    float inv = 1.0f / fmaxf(sqrtf(ss), 1e-6f);
    uint2* o = (uint2*)(outb + (size_t)row * DD);
#pragma unroll
    for (int c = 0; c < 4; ++c) {
        uint2 p;
        p.x = (unsigned)f2bf(x[c].x * inv) | ((unsigned)f2bf(x[c].y * inv) << 16);
        p.y = (unsigned)f2bf(x[c].z * inv) | ((unsigned)f2bf(x[c].w * inv) << 16);
        o[lane + 32 * c] = p;
    }
}

// ------------------------- Pass 2: fused WMMA head -------------------------
// Block = 128 threads (4 waves): blockIdx.x = a, blockIdx.y = group of 4 b's.
// Wave w owns b = 4*blockIdx.y + w.
__global__ __launch_bounds__(128) void drl_head_wmma(const __bf16* __restrict__ Tbf,
                                                     const __bf16* __restrict__ Vbf,
                                                     const float* __restrict__ mask,
                                                     float* __restrict__ out) {
    extern __shared__ char smem[];
    __bf16* sT = (__bf16*)smem;            // 32 rows * STR
    __bf16* sV = sT + T_TOK * STR;         // 64 rows * STR
    float*  sM = (float*)(sV + 4 * V_FRM * STR);

    const int a = blockIdx.x, bg = blockIdx.y, tid = threadIdx.x;

    // Cooperative LDS staging (uint4 = 8 halfs per op), padded rows.
    const uint4* gT = (const uint4*)(Tbf + (size_t)a * T_TOK * DD);
    for (int c = tid; c < T_TOK * (DD / 8); c += 128) {
        int row = c >> 6, col = c & 63;
        *(uint4*)(sT + row * STR + col * 8) = gT[row * 64 + col];
    }
    const uint4* gV = (const uint4*)(Vbf + (size_t)bg * 4 * V_FRM * DD);
    for (int c = tid; c < 4 * V_FRM * (DD / 8); c += 128) {
        int row = c >> 6, col = c & 63;
        *(uint4*)(sV + row * STR + col * 8) = gV[row * 64 + col];
    }
    if (tid < T_TOK) sM[tid] = mask[a * T_TOK + tid];
    __syncthreads();

    const int wave = tid >> 5, lane = tid & 31;
    const int n = lane & 15, h = lane >> 4;          // C layout: col=n, rows 8h+i

    const __bf16* arow0 = sT + n * STR;              // A rows 0..15  (m = lane&15)
    const __bf16* arow1 = sT + (16 + n) * STR;       // A rows 16..31
    const __bf16* brow  = sV + (wave * V_FRM + n) * STR; // B = videoT, K-contiguous

    v8f acc0 = {0.f, 0.f, 0.f, 0.f, 0.f, 0.f, 0.f, 0.f};
    v8f acc1 = {0.f, 0.f, 0.f, 0.f, 0.f, 0.f, 0.f, 0.f};

#pragma unroll 4
    for (int kc = 0; kc < DD; kc += 32) {
        union { uint4 u[2]; v16bf v; } fa0, fa1, fb;
        const int o1 = kc + 8 * h, o2 = kc + 16 + 8 * h;
        fa0.u[0] = *(const uint4*)(arow0 + o1); fa0.u[1] = *(const uint4*)(arow0 + o2);
        fa1.u[0] = *(const uint4*)(arow1 + o1); fa1.u[1] = *(const uint4*)(arow1 + o2);
        fb.u[0]  = *(const uint4*)(brow  + o1); fb.u[1]  = *(const uint4*)(brow  + o2);
        acc0 = __builtin_amdgcn_wmma_f32_16x16x32_bf16(false, fa0.v, false, fb.v,
                                                       (short)0, acc0, false, false);
        acc1 = __builtin_amdgcn_wmma_f32_16x16x32_bf16(false, fa1.v, false, fb.v,
                                                       (short)0, acc1, false, false);
    }

    // ---- Epilogue: lane holds logits L[t][n] for t = 8h+i (acc0), 16+8h+i (acc1)
    const float NEG = -1e30f;
    float L0[8], L1[8];
#pragma unroll
    for (int i = 0; i < 8; ++i) {
        L0[i] = acc0[i] * sM[8 * h + i];
        L1[i] = acc1[i] * sM[16 + 8 * h + i];
    }

    // t2v level 1: softmax over v (16 lanes within each half: xor 1,2,4,8)
    float t2v0[8], t2v1[8];
#pragma unroll
    for (int i = 0; i < 8; ++i) {
        float x = L0[i], r = x;
        r = fmaxf(r, xsh(r, 1)); r = fmaxf(r, xsh(r, 2));
        r = fmaxf(r, xsh(r, 4)); r = fmaxf(r, xsh(r, 8));
        float e = __expf(TAUF * (x - r));
        float se = e, sl = x * e;
        se += xsh(se, 1); se += xsh(se, 2); se += xsh(se, 4); se += xsh(se, 8);
        sl += xsh(sl, 1); sl += xsh(sl, 2); sl += xsh(sl, 4); sl += xsh(sl, 8);
        t2v0[i] = sl / se;
        x = L1[i]; r = x;
        r = fmaxf(r, xsh(r, 1)); r = fmaxf(r, xsh(r, 2));
        r = fmaxf(r, xsh(r, 4)); r = fmaxf(r, xsh(r, 8));
        e = __expf(TAUF * (x - r));
        se = e; sl = x * e;
        se += xsh(se, 1); se += xsh(se, 2); se += xsh(se, 4); se += xsh(se, 8);
        sl += xsh(sl, 1); sl += xsh(sl, 2); sl += xsh(sl, 4); sl += xsh(sl, 8);
        t2v1[i] = sl / se;
    }

    // v2t level 1: masked softmax over t (intra-lane over 16 regs, then xor 16)
    float cm = NEG;
#pragma unroll
    for (int i = 0; i < 8; ++i) {
        cm = fmaxf(cm, (L0[i] == 0.f) ? NEG : TAUF * L0[i]);
        cm = fmaxf(cm, (L1[i] == 0.f) ? NEG : TAUF * L1[i]);
    }
    cm = fmaxf(cm, xsh(cm, 16));
    float se_c = 0.f, sl_c = 0.f;
#pragma unroll
    for (int i = 0; i < 8; ++i) {
        float e0 = (L0[i] == 0.f) ? 0.f : __expf(TAUF * L0[i] - cm);
        float e1 = (L1[i] == 0.f) ? 0.f : __expf(TAUF * L1[i] - cm);
        se_c += e0 + e1;
        sl_c += L0[i] * e0 + L1[i] * e1;
    }
    se_c += xsh(se_c, 16);
    sl_c += xsh(sl_c, 16);
    float v2t = sl_c / se_c;   // per column n, replicated on lanes n, n+16

    // t2v level 2: masked softmax-pool over all 32 t
    float gm = NEG;
#pragma unroll
    for (int i = 0; i < 8; ++i) {
        gm = fmaxf(gm, (t2v0[i] == 0.f) ? NEG : TAUF * t2v0[i]);
        gm = fmaxf(gm, (t2v1[i] == 0.f) ? NEG : TAUF * t2v1[i]);
    }
    gm = fmaxf(gm, xsh(gm, 16));
    float s2 = 0.f, sv2 = 0.f;
#pragma unroll
    for (int i = 0; i < 8; ++i) {
        float e0 = (t2v0[i] == 0.f) ? 0.f : __expf(TAUF * t2v0[i] - gm);
        float e1 = (t2v1[i] == 0.f) ? 0.f : __expf(TAUF * t2v1[i] - gm);
        s2  += e0 + e1;
        sv2 += t2v0[i] * e0 + t2v1[i] * e1;
    }
    s2  += xsh(s2, 16);
    sv2 += xsh(sv2, 16);
    float t2vF = sv2 / s2;

    // v2t level 2: softmax-pool over 16 v (xor 1,2,4,8 within half)
    float m2 = TAUF * v2t;
    m2 = fmaxf(m2, xsh(m2, 1)); m2 = fmaxf(m2, xsh(m2, 2));
    m2 = fmaxf(m2, xsh(m2, 4)); m2 = fmaxf(m2, xsh(m2, 8));
    float e2 = __expf(TAUF * v2t - m2);
    float ssum = e2, svsum = v2t * e2;
    ssum  += xsh(ssum, 1);  ssum  += xsh(ssum, 2);  ssum  += xsh(ssum, 4);  ssum  += xsh(ssum, 8);
    svsum += xsh(svsum, 1); svsum += xsh(svsum, 2); svsum += xsh(svsum, 4); svsum += xsh(svsum, 8);
    float v2tF = svsum / ssum;

    if (lane == 0)
        out[(size_t)a * B_N + (bg * 4 + wave)] = 0.5f * (t2vF + v2tF);
}

// ------------------------------- Launcher ----------------------------------
extern "C" void kernel_launch(void* const* d_in, const int* in_sizes, int n_in,
                              void* d_out, int out_size, void* d_ws, size_t ws_size,
                              hipStream_t stream) {
    (void)in_sizes; (void)n_in; (void)out_size; (void)ws_size;
    const float* text  = (const float*)d_in[0];   // [256,32,512] f32
    const float* video = (const float*)d_in[1];   // [256,16,512] f32
    const float* mask  = (const float*)d_in[2];   // [256,32]     f32
    float*       out   = (float*)d_out;           // [256,256]    f32

    unsigned short* wsT = (unsigned short*)d_ws;                       // 8192*512 bf16
    unsigned short* wsV = wsT + (size_t)A_N * T_TOK * DD;              // 4096*512 bf16

    // Pass 1: normalize + convert (8 waves/block, 1 row/wave)
    drl_norm_bf16<<<(A_N * T_TOK) / 8, 256, 0, stream>>>(text,  wsT, A_N * T_TOK);
    drl_norm_bf16<<<(B_N * V_FRM) / 8, 256, 0, stream>>>(video, wsV, B_N * V_FRM);

    // Pass 2: fused GEMM + pooling. LDS: (32+64)*520 halfs + 32 floats ~= 97.6KB
    size_t lds = (size_t)(T_TOK + 4 * V_FRM) * STR * sizeof(unsigned short)
               + T_TOK * sizeof(float);
    dim3 grid(A_N, B_N / 4);
    drl_head_wmma<<<grid, 128, lds, stream>>>((const __bf16*)wsT, (const __bf16*)wsV,
                                              mask, out);
}
